// PointNetSetAbstraction_85117661872572
// MI455X (gfx1250) — matrix operand order repaired and checked by reading, not compiled, not run on hardware
//
#include <hip/hip_runtime.h>

// PointNet Set Abstraction for MI455X (gfx1250, wave32, WMMA).
// Workspace requirement: ~150 MB (knn idx 2MB + y0/y1 f16 67MB each + max/min 16MB + partials ~6MB).

#define BATCH 16
#define NPTS  4096
#define DFEAT 64
#define S     1024
#define KNN   32
#define C0    67
#define C0P   96          // 67 padded to 3 chunks of 32
#define ROWS  ((size_t)BATCH * S * KNN)   // 524288

typedef __attribute__((ext_vector_type(16))) _Float16 v16h;
typedef __attribute__((ext_vector_type(8)))  float    v8f;

// ---- WMMA fragment loaders (CDNA5 16-bit layouts, ISA 7.12.2) -------------
// A 16x32 (MxK), row-major LDS tile with leading dim ldk (halfs).
__device__ __forceinline__ v16h load_a_frag(const _Float16* tA, int ldk, int kc, int lane) {
  int m  = lane & 15;
  int kb = kc + ((lane >> 4) << 3);     // +8 for upper half-wave
  const _Float16* rp = tA + m * ldk;
  v16h a;
#pragma unroll
  for (int e = 0; e < 8; ++e) a[e]     = rp[kb + e];
#pragma unroll
  for (int e = 0; e < 8; ++e) a[8 + e] = rp[kb + 16 + e];
  return a;
}
// B 32x16 (KxN) built from weights W (row-major [Cout][ldc]); B[k][n] = W[n0*16+n][kc+k].
__device__ __forceinline__ v16h load_b_frag(const _Float16* W, int ldc, int n0, int kc, int lane) {
  int n  = lane & 15;
  int kb = kc + ((lane >> 4) << 3);     // +8 for upper half-wave
  const _Float16* cp = W + (size_t)(n0 * 16 + n) * ldc + kb;
  v16h b;
#pragma unroll
  for (int e = 0; e < 8; ++e) { b[e] = cp[e]; b[8 + e] = cp[16 + e]; }
  return b;
}

// ---- 1. Farthest point sampling: one workgroup per batch, LDS-resident ----
__global__ __launch_bounds__(1024) void fps_kernel(const float* __restrict__ pts,
                                                   float* __restrict__ centers_xyz) {
  int b = blockIdx.x;
  __shared__ float px[NPTS], py[NPTS], pz[NPTS], dd[NPTS];
  __shared__ float wv[32];
  __shared__ int   wi[32];
  __shared__ float cc[3];
  int tid = threadIdx.x, lane = tid & 31, w = tid >> 5;
  for (int i = tid; i < NPTS; i += 1024) {
    const float* p = pts + ((size_t)b * NPTS + i) * 3;
    px[i] = p[0]; py[i] = p[1]; pz[i] = p[2];
  }
  __syncthreads();
  float cx = px[0], cy = py[0], cz = pz[0];
  for (int i = tid; i < NPTS; i += 1024) {
    float dx = px[i] - cx, dy = py[i] - cy, dz = pz[i] - cz;
    dd[i] = dx * dx + dy * dy + dz * dz;
  }
  if (tid == 0) {
    float* o = centers_xyz + (size_t)b * S * 3;
    o[0] = cx; o[1] = cy; o[2] = cz;
  }
  for (int it = 1; it < S; ++it) {
    __syncthreads();
    float bv = -1.0f; int bi = 0;
    for (int i = tid; i < NPTS; i += 1024) {          // ascending -> lowest idx on tie
      float v = dd[i];
      if (v > bv) { bv = v; bi = i; }
    }
#pragma unroll
    for (int o = 16; o > 0; o >>= 1) {
      float ov = __shfl_down(bv, o, 32);
      int   oi = __shfl_down(bi, o, 32);
      if (ov > bv || (ov == bv && oi < bi)) { bv = ov; bi = oi; }
    }
    if (lane == 0) { wv[w] = bv; wi[w] = bi; }
    __syncthreads();
    if (w == 0) {
      bv = wv[lane]; bi = wi[lane];
#pragma unroll
      for (int o = 16; o > 0; o >>= 1) {
        float ov = __shfl_down(bv, o, 32);
        int   oi = __shfl_down(bi, o, 32);
        if (ov > bv || (ov == bv && oi < bi)) { bv = ov; bi = oi; }
      }
      if (lane == 0) { cc[0] = px[bi]; cc[1] = py[bi]; cc[2] = pz[bi]; }
    }
    __syncthreads();
    float ccx = cc[0], ccy = cc[1], ccz = cc[2];
    if (tid == 0) {
      float* o = centers_xyz + ((size_t)b * S + it) * 3;
      o[0] = ccx; o[1] = ccy; o[2] = ccz;
    }
    for (int i = tid; i < NPTS; i += 1024) {
      float dx = px[i] - ccx, dy = py[i] - ccy, dz = pz[i] - ccz;
      float nd = dx * dx + dy * dy + dz * dz;
      dd[i] = fminf(dd[i], nd);
    }
  }
}

// ---- 2. KNN: one workgroup per center, 32 iterative argmins over LDS ------
__global__ __launch_bounds__(256) void knn_kernel(const float* __restrict__ pts,
                                                  const float* __restrict__ centers_xyz,
                                                  int* __restrict__ knn) {
  int cs = blockIdx.x;               // b*S + s
  int b  = cs >> 10;
  __shared__ float sD[NPTS];
  __shared__ float wv[8];
  __shared__ int   wi[8];
  int tid = threadIdx.x, lane = tid & 31, w = tid >> 5;
  float cx = centers_xyz[(size_t)cs * 3 + 0];
  float cy = centers_xyz[(size_t)cs * 3 + 1];
  float cz = centers_xyz[(size_t)cs * 3 + 2];
  for (int i = tid; i < NPTS; i += 256) {
    const float* p = pts + ((size_t)b * NPTS + i) * 3;
    float dx = p[0] - cx, dy = p[1] - cy, dz = p[2] - cz;
    sD[i] = dx * dx + dy * dy + dz * dz;
  }
  __syncthreads();
  for (int k = 0; k < KNN; ++k) {
    float bv = 3.402823466e38f; int bi = 0;
    for (int i = tid; i < NPTS; i += 256) {
      float v = sD[i];
      if (v < bv) { bv = v; bi = i; }
    }
#pragma unroll
    for (int o = 16; o > 0; o >>= 1) {
      float ov = __shfl_down(bv, o, 32);
      int   oi = __shfl_down(bi, o, 32);
      if (ov < bv || (ov == bv && oi < bi)) { bv = ov; bi = oi; }
    }
    if (lane == 0) { wv[w] = bv; wi[w] = bi; }
    __syncthreads();
    if (tid == 0) {
      float fb = wv[0]; int fi = wi[0];
      for (int ww = 1; ww < 8; ++ww)
        if (wv[ww] < fb || (wv[ww] == fb && wi[ww] < fi)) { fb = wv[ww]; fi = wi[ww]; }
      knn[(size_t)cs * KNN + k] = fi;
      sD[fi] = 3.402823466e38f;
    }
    __syncthreads();
  }
}

// ---- 3. Layer 0: gather + WMMA (67->64), f16 pre-BN out + per-WG stats ----
__global__ __launch_bounds__(256) void layer0_kernel(
    const float* __restrict__ pts, const float* __restrict__ feat,
    const float* __restrict__ centers_xyz, const int* __restrict__ knn,
    const float* __restrict__ W0, _Float16* __restrict__ y0, float* __restrict__ gp0) {
  __shared__ _Float16 sW[64 * C0P];        // 12 KB
  __shared__ _Float16 sA[8][16 * C0P];     // 24 KB
  __shared__ float    sPart[8][64][2];     // 4 KB
  int tid = threadIdx.x, lane = tid & 31, w = tid >> 5;
  for (int i = tid; i < 64 * C0P; i += 256) {
    int oc = i / C0P, c = i % C0P;
    sW[i] = (_Float16)((c < C0) ? W0[oc * C0 + c] : 0.0f);
  }
  int rowBase = blockIdx.x * 128 + w * 16;
  for (int r = 0; r < 16; ++r) {
    int row = rowBase + r;
    int bs  = row >> 5;                    // /KNN
    int b   = bs >> 10;                    // /S
    int idx = knn[row];
    const float* prow = pts + ((size_t)b * NPTS + idx) * 3;
    const float* frow = feat + ((size_t)b * NPTS + idx) * DFEAT;
    const float* crow = centers_xyz + (size_t)bs * 3;
    for (int c = lane; c < C0P; c += 32) {
      float v;
      if (c < 3)        v = prow[c] - crow[c];
      else if (c < C0)  v = frow[c - 3];
      else              v = 0.0f;
      sA[w][r * C0P + c] = (_Float16)v;
    }
  }
  __syncthreads();
  v8f acc[4] = {};
  for (int kc = 0; kc < C0P; kc += 32) {
    v16h a = load_a_frag(&sA[w][0], C0P, kc, lane);
#pragma unroll
    for (int n0 = 0; n0 < 4; ++n0) {
      v16h bf = load_b_frag(sW, C0P, n0, kc, lane);
      acc[n0] = __builtin_amdgcn_wmma_f32_16x16x32_f16(false, a, false, bf, (short)0, acc[n0], false, false);
    }
  }
  int mb = (lane >> 4) << 3, nc = lane & 15;
#pragma unroll
  for (int n0 = 0; n0 < 4; ++n0) {
    float s = 0.0f, s2 = 0.0f;
#pragma unroll
    for (int i = 0; i < 8; ++i) {
      float v = acc[n0][i];
      y0[(size_t)(rowBase + mb + i) * 64 + n0 * 16 + nc] = (_Float16)v;
      s += v; s2 += v * v;
    }
    s  += __shfl_xor(s, 16, 32);
    s2 += __shfl_xor(s2, 16, 32);
    if (lane < 16) { sPart[w][n0 * 16 + lane][0] = s; sPart[w][n0 * 16 + lane][1] = s2; }
  }
  __syncthreads();
  if (tid < 64) {
    float s = 0.0f, s2 = 0.0f;
    for (int ww = 0; ww < 8; ++ww) { s += sPart[ww][tid][0]; s2 += sPart[ww][tid][1]; }
    gp0[((size_t)blockIdx.x * 64 + tid) * 2 + 0] = s;
    gp0[((size_t)blockIdx.x * 64 + tid) * 2 + 1] = s2;
  }
}

// ---- BN stats: deterministic fixed-order reduce over per-WG partials ------
__global__ __launch_bounds__(128) void stats_kernel(
    const float* __restrict__ gp, int nwg, int C,
    const float* __restrict__ gamma, const float* __restrict__ beta,
    float* __restrict__ sb, float invCount) {
  int c = threadIdx.x;
  if (c >= C) return;
  float s = 0.0f, s2 = 0.0f;
  for (int i = 0; i < nwg; ++i) {
    s  += gp[((size_t)i * C + c) * 2 + 0];
    s2 += gp[((size_t)i * C + c) * 2 + 1];
  }
  float m   = s * invCount;
  float var = fmaxf(s2 * invCount - m * m, 0.0f);
  float sc  = gamma[c] * rsqrtf(var + 1e-5f);
  sb[c * 2 + 0] = sc;
  sb[c * 2 + 1] = beta[c] - m * sc;
}

// ---- 4. Layer 1: BN+ReLU(y0) -> WMMA (64->64) -----------------------------
__global__ __launch_bounds__(256) void layer1_kernel(
    const _Float16* __restrict__ y0, const float* __restrict__ W1,
    const float* __restrict__ sb0, _Float16* __restrict__ y1, float* __restrict__ gp1) {
  __shared__ _Float16 sW[64 * 64];
  __shared__ _Float16 sA[8][16 * 64];
  __shared__ float    sPart[8][64][2];
  __shared__ float    sScale[64], sBias[64];
  int tid = threadIdx.x, lane = tid & 31, w = tid >> 5;
  for (int i = tid; i < 64 * 64; i += 256) sW[i] = (_Float16)W1[i];
  if (tid < 64) { sScale[tid] = sb0[tid * 2]; sBias[tid] = sb0[tid * 2 + 1]; }
  __syncthreads();
  int rowBase = blockIdx.x * 128 + w * 16;
  for (int r = 0; r < 16; ++r) {
    const _Float16* yr = y0 + (size_t)(rowBase + r) * 64;
    for (int c = lane; c < 64; c += 32) {
      float v = fmaxf((float)yr[c] * sScale[c] + sBias[c], 0.0f);
      sA[w][r * 64 + c] = (_Float16)v;
    }
  }
  __syncthreads();
  v8f acc[4] = {};
  for (int kc = 0; kc < 64; kc += 32) {
    v16h a = load_a_frag(&sA[w][0], 64, kc, lane);
#pragma unroll
    for (int n0 = 0; n0 < 4; ++n0) {
      v16h bf = load_b_frag(sW, 64, n0, kc, lane);
      acc[n0] = __builtin_amdgcn_wmma_f32_16x16x32_f16(false, a, false, bf, (short)0, acc[n0], false, false);
    }
  }
  int mb = (lane >> 4) << 3, nc = lane & 15;
#pragma unroll
  for (int n0 = 0; n0 < 4; ++n0) {
    float s = 0.0f, s2 = 0.0f;
#pragma unroll
    for (int i = 0; i < 8; ++i) {
      float v = acc[n0][i];
      y1[(size_t)(rowBase + mb + i) * 64 + n0 * 16 + nc] = (_Float16)v;
      s += v; s2 += v * v;
    }
    s  += __shfl_xor(s, 16, 32);
    s2 += __shfl_xor(s2, 16, 32);
    if (lane < 16) { sPart[w][n0 * 16 + lane][0] = s; sPart[w][n0 * 16 + lane][1] = s2; }
  }
  __syncthreads();
  if (tid < 64) {
    float s = 0.0f, s2 = 0.0f;
    for (int ww = 0; ww < 8; ++ww) { s += sPart[ww][tid][0]; s2 += sPart[ww][tid][1]; }
    gp1[((size_t)blockIdx.x * 64 + tid) * 2 + 0] = s;
    gp1[((size_t)blockIdx.x * 64 + tid) * 2 + 1] = s2;
  }
}

// ---- 5. Layer 2: WMMA (64->128) with in-register K-max/min per center -----
__global__ __launch_bounds__(256) void layer2_kernel(
    const _Float16* __restrict__ y1, const float* __restrict__ W2,
    const float* __restrict__ sb1, float* __restrict__ ymax, float* __restrict__ ymin,
    float* __restrict__ gp2) {
  __shared__ _Float16 sW[128 * 64];        // 16 KB
  __shared__ _Float16 sA[8][32 * 64];      // 32 KB
  __shared__ float    sPart[8][128][2];    // 8 KB
  __shared__ float    sScale[64], sBias[64];
  int tid = threadIdx.x, lane = tid & 31, w = tid >> 5;
  for (int i = tid; i < 128 * 64; i += 256) sW[i] = (_Float16)W2[i];
  if (tid < 64) { sScale[tid] = sb1[tid * 2]; sBias[tid] = sb1[tid * 2 + 1]; }
  __syncthreads();
  int bs = blockIdx.x * 8 + w;             // one center per wave
  int rowBase = bs * KNN;
  for (int r = 0; r < 32; ++r) {
    const _Float16* yr = y1 + (size_t)(rowBase + r) * 64;
    for (int c = lane; c < 64; c += 32) {
      float v = fmaxf((float)yr[c] * sScale[c] + sBias[c], 0.0f);
      sA[w][r * 64 + c] = (_Float16)v;
    }
  }
  __syncthreads();
  v16h a[2][2];
#pragma unroll
  for (int rt = 0; rt < 2; ++rt)
#pragma unroll
    for (int q = 0; q < 2; ++q)
      a[rt][q] = load_a_frag(&sA[w][rt * 16 * 64], 64, q * 32, lane);
#pragma unroll
  for (int n0 = 0; n0 < 8; ++n0) {
    v8f acc0 = {}, acc1 = {};
#pragma unroll
    for (int q = 0; q < 2; ++q) {
      v16h bf = load_b_frag(sW, 64, n0, q * 32, lane);
      acc0 = __builtin_amdgcn_wmma_f32_16x16x32_f16(false, a[0][q], false, bf, (short)0, acc0, false, false);
      acc1 = __builtin_amdgcn_wmma_f32_16x16x32_f16(false, a[1][q], false, bf, (short)0, acc1, false, false);
    }
    float mx = -3.402823466e38f, mn = 3.402823466e38f, s = 0.0f, s2 = 0.0f;
#pragma unroll
    for (int i = 0; i < 8; ++i) {
      float v0 = acc0[i], v1 = acc1[i];
      mx = fmaxf(mx, fmaxf(v0, v1));
      mn = fminf(mn, fminf(v0, v1));
      s += v0 + v1; s2 += v0 * v0 + v1 * v1;
    }
    mx = fmaxf(mx, __shfl_xor(mx, 16, 32));
    mn = fminf(mn, __shfl_xor(mn, 16, 32));
    s  += __shfl_xor(s, 16, 32);
    s2 += __shfl_xor(s2, 16, 32);
    if (lane < 16) {
      ymax[(size_t)bs * 128 + n0 * 16 + lane] = mx;
      ymin[(size_t)bs * 128 + n0 * 16 + lane] = mn;
      sPart[w][n0 * 16 + lane][0] = s;
      sPart[w][n0 * 16 + lane][1] = s2;
    }
  }
  __syncthreads();
  if (tid < 128) {
    float s = 0.0f, s2 = 0.0f;
    for (int ww = 0; ww < 8; ++ww) { s += sPart[ww][tid][0]; s2 += sPart[ww][tid][1]; }
    gp2[((size_t)blockIdx.x * 128 + tid) * 2 + 0] = s;
    gp2[((size_t)blockIdx.x * 128 + tid) * 2 + 1] = s2;
  }
}

// ---- 6. Finalize: max_k(relu(bn(y))) via monotonicity (max if scale>=0) ---
__global__ __launch_bounds__(256) void finalize_kernel(
    const float* __restrict__ ymax, const float* __restrict__ ymin,
    const float* __restrict__ sb2, float* __restrict__ outp) {
  size_t i = (size_t)blockIdx.x * 256 + threadIdx.x;   // over B*S*128
  int c = (int)(i & 127);
  float sc = sb2[c * 2], bi = sb2[c * 2 + 1];
  float v = (sc >= 0.0f) ? ymax[i] : ymin[i];
  outp[i] = fmaxf(sc * v + bi, 0.0f);
}

extern "C" void kernel_launch(void* const* d_in, const int* in_sizes, int n_in,
                              void* d_out, int out_size, void* d_ws, size_t ws_size,
                              hipStream_t stream) {
  (void)in_sizes; (void)n_in; (void)out_size; (void)ws_size;
  const float* pts  = (const float*)d_in[0];
  const float* feat = (const float*)d_in[1];
  const float* W0   = (const float*)d_in[2];
  const float* g0   = (const float*)d_in[3];
  const float* b0   = (const float*)d_in[4];
  const float* W1   = (const float*)d_in[5];
  const float* g1   = (const float*)d_in[6];
  const float* b1   = (const float*)d_in[7];
  const float* W2   = (const float*)d_in[8];
  const float* g2   = (const float*)d_in[9];
  const float* b2   = (const float*)d_in[10];

  float* out          = (float*)d_out;
  float* centers_xyz  = out;                          // 16*1024*3
  float* center_feat  = out + (size_t)BATCH * S * 3;  // 16*1024*128

  char* ws = (char*)d_ws;
  size_t o = 0;
  int*      knn  = (int*)(ws + o);      o += ROWS * sizeof(int);
  _Float16* y0   = (_Float16*)(ws + o); o += ROWS * 64 * sizeof(_Float16);
  _Float16* y1   = (_Float16*)(ws + o); o += ROWS * 64 * sizeof(_Float16);
  float*    ymax = (float*)(ws + o);    o += (size_t)BATCH * S * 128 * sizeof(float);
  float*    ymin = (float*)(ws + o);    o += (size_t)BATCH * S * 128 * sizeof(float);
  float*    gp0  = (float*)(ws + o);    o += (size_t)4096 * 64 * 2 * sizeof(float);
  float*    gp1  = (float*)(ws + o);    o += (size_t)4096 * 64 * 2 * sizeof(float);
  float*    gp2  = (float*)(ws + o);    o += (size_t)2048 * 128 * 2 * sizeof(float);
  float*    sb0  = (float*)(ws + o);    o += 128 * sizeof(float);
  float*    sb1  = (float*)(ws + o);    o += 128 * sizeof(float);
  float*    sb2  = (float*)(ws + o);    o += 256 * sizeof(float);

  const float invN = 1.0f / (float)ROWS;

  fps_kernel<<<BATCH, 1024, 0, stream>>>(pts, centers_xyz);
  knn_kernel<<<BATCH * S, 256, 0, stream>>>(pts, centers_xyz, knn);
  layer0_kernel<<<4096, 256, 0, stream>>>(pts, feat, centers_xyz, knn, W0, y0, gp0);
  stats_kernel<<<1, 128, 0, stream>>>(gp0, 4096, 64, g0, b0, sb0, invN);
  layer1_kernel<<<4096, 256, 0, stream>>>(y0, W1, sb0, y1, gp1);
  stats_kernel<<<1, 128, 0, stream>>>(gp1, 4096, 64, g1, b1, sb1, invN);
  layer2_kernel<<<2048, 256, 0, stream>>>(y1, W2, sb1, ymax, ymin, gp2);
  stats_kernel<<<1, 128, 0, stream>>>(gp2, 2048, 128, g2, b2, sb2, invN);
  finalize_kernel<<<(BATCH * S * 128) / 256, 256, 0, stream>>>(ymax, ymin, sb2, center_feat);
}
